// MoE_72971494359126
// MI455X (gfx1250) — compile-verified, matrix operand order checked by reading, not run on hardware
//
#include <hip/hip_runtime.h>
#include <cstdint>
#include <cstddef>

// ---------------- problem constants (from reference) ----------------
#define E_      16
#define KTOP    2
#define D_      1024
#define H_      1408
#define N_      8192            // B*T = 4*2048
#define CAP_    1280            // ceil(1.25 * K * N / E)
#define MROWS_  (N_ + E_*CAP_)  // 28672 = shared rows + expert slot rows

// ---------------- types ----------------
typedef float  v8f   __attribute__((ext_vector_type(8)));
typedef __bf16 v16bf __attribute__((ext_vector_type(16)));
typedef __bf16 bf8   __attribute__((ext_vector_type(8)));
typedef __bf16 bf4   __attribute__((ext_vector_type(4)));

static __device__ __forceinline__ __bf16 f2bf(float f) {
  unsigned u = __builtin_bit_cast(unsigned, f);
  unsigned r = u + 0x7FFFu + ((u >> 16) & 1u);   // round-to-nearest-even
  unsigned short s = (unsigned short)(r >> 16);
  return __builtin_bit_cast(__bf16, s);
}
static __device__ __forceinline__ float bf2f(__bf16 b) {
  unsigned short s = __builtin_bit_cast(unsigned short, b);
  unsigned u = ((unsigned)s) << 16;
  return __builtin_bit_cast(float, u);
}

// CDNA5 async global->LDS copy (ASYNCcnt path). INST_OFFSET is added to both
// the LDS and the global address (ISA 08_async_tensor §4.4).
static __device__ __forceinline__ void async_copy_b128(unsigned lds, const void* g) {
  asm volatile("global_load_async_to_lds_b128 %0, %1, off"
               :: "v"(lds), "v"(g) : "memory");
}
static __device__ __forceinline__ void async_copy_b128_o16(unsigned lds, const void* g) {
  asm volatile("global_load_async_to_lds_b128 %0, %1, off offset:16"
               :: "v"(lds), "v"(g) : "memory");
}
static __device__ __forceinline__ void wait_async0() {
  asm volatile("s_wait_asynccnt 0x0" ::: "memory");
}

// ---------------- 1) router: logits, softmax, top-2, gates ----------------
__global__ __launch_bounds__(256) void router_kernel(
    const float* __restrict__ x, const float* __restrict__ rw,
    float* __restrict__ probs, int* __restrict__ top1,
    int* __restrict__ eidx, float* __restrict__ gate)
{
  const int wave = threadIdx.x >> 5;
  const int lane = threadIdx.x & 31;
  const int n = blockIdx.x * 8 + wave;

  float xr[32];
  const float* xp = x + (size_t)n * D_;
  #pragma unroll
  for (int j = 0; j < 32; ++j) xr[j] = xp[j * 32 + lane];

  float myLogit = -3.0e38f;
  for (int e = 0; e < E_; ++e) {
    const float* wp = rw + (size_t)e * D_;
    float p = 0.f;
    #pragma unroll
    for (int j = 0; j < 32; ++j) p += xr[j] * wp[j * 32 + lane];
    #pragma unroll
    for (int m = 16; m >= 1; m >>= 1) p += __shfl_xor(p, m, 32);
    if (lane == e) myLogit = p;     // lane e (0..15) keeps logit_e
  }

  // softmax over the 16-lane group
  float mx = myLogit;
  #pragma unroll
  for (int m = 8; m >= 1; m >>= 1) mx = fmaxf(mx, __shfl_xor(mx, m, 16));
  float pe = __expf(myLogit - mx);
  float s = pe;
  #pragma unroll
  for (int m = 8; m >= 1; m >>= 1) s += __shfl_xor(s, m, 16);
  const float prob = pe / s;
  if (lane < 16) probs[(size_t)n * E_ + lane] = prob;

  // top-1 (ties -> lowest index, matches lax.top_k)
  float v1 = prob; int i1 = lane & 15;
  #pragma unroll
  for (int m = 8; m >= 1; m >>= 1) {
    float ov = __shfl_xor(v1, m, 16);
    int   oi = __shfl_xor(i1, m, 16);
    if (ov > v1 || (ov == v1 && oi < i1)) { v1 = ov; i1 = oi; }
  }
  // top-2
  float v2 = ((lane & 15) == i1) ? -1.f : prob; int i2 = lane & 15;
  #pragma unroll
  for (int m = 8; m >= 1; m >>= 1) {
    float ov = __shfl_xor(v2, m, 16);
    int   oi = __shfl_xor(i2, m, 16);
    if (ov > v2 || (ov == v2 && oi < i2)) { v2 = ov; i2 = oi; }
  }
  if (lane == 0) {
    const float den = v1 + v2 + 1e-9f;
    eidx[n * 2 + 0] = i1;       eidx[n * 2 + 1] = i2;
    gate[n * 2 + 0] = v1 / den; gate[n * 2 + 1] = v2 / den;
    top1[n] = i1;
  }
}

// ---------------- 2) deterministic capacity dispatch (stable order) --------
__global__ __launch_bounds__(256) void dispatch_kernel(
    const int* __restrict__ eidx, int* __restrict__ slot_of,
    int* __restrict__ slot_token)
{
  for (int s = threadIdx.x; s < E_ * CAP_; s += 256) slot_token[s] = -1;
  __syncthreads();
  if (threadIdx.x < E_) {
    const int e = threadIdx.x;
    int cnt = 0;
    for (int f = 0; f < N_ * KTOP; ++f) {   // token-major == stable argsort order
      if (eidx[f] == e) {
        if (cnt < CAP_) {
          slot_of[f] = e * CAP_ + cnt;
          slot_token[e * CAP_ + cnt] = f >> 1;   // token id
        } else {
          slot_of[f] = -1;                       // dropped (over capacity)
        }
        ++cnt;
      }
    }
  }
}

// ---------------- 3) gather into unified bf16 A matrix --------------------
__global__ __launch_bounds__(256) void gather_kernel(
    const float* __restrict__ x, const int* __restrict__ slot_token,
    __bf16* __restrict__ Abuf)
{
  const int row = blockIdx.x;
  const int tok = (row < N_) ? row : slot_token[row - N_];
  const int c = threadIdx.x * 4;
  const __bf16 z = __builtin_bit_cast(__bf16, (unsigned short)0);
  bf4 o = {z, z, z, z};
  if (tok >= 0) {
    const float4 v = *(const float4*)(x + (size_t)tok * D_ + c);
    o[0] = f2bf(v.x); o[1] = f2bf(v.y); o[2] = f2bf(v.z); o[3] = f2bf(v.w);
  }
  *(bf4*)(Abuf + (size_t)row * D_ + c) = o;
}

// ------- 3b) one-time weight convert+transpose: fp32 [K,N] -> bf16 [N,K] ---
__global__ __launch_bounds__(256) void wtrans_kernel(
    const float* __restrict__ W, __bf16* __restrict__ Wt, int Kd, int Nd)
{
  __shared__ float t[32][33];
  const float* Wg = W + (size_t)blockIdx.z * Kd * Nd;
  __bf16* Wtg = Wt + (size_t)blockIdx.z * Kd * Nd;
  const int k0 = blockIdx.x * 32, n0 = blockIdx.y * 32;
  const int tx = threadIdx.x & 31, ty0 = threadIdx.x >> 5;
  #pragma unroll
  for (int i = 0; i < 4; ++i) {
    const int ty = ty0 + i * 8;
    t[ty][tx] = Wg[(size_t)(k0 + ty) * Nd + n0 + tx];
  }
  __syncthreads();
  #pragma unroll
  for (int i = 0; i < 4; ++i) {
    const int ty = ty0 + i * 8;
    Wtg[(size_t)(n0 + ty) * Kd + k0 + tx] = f2bf(t[tx][ty]);
  }
}

// ---------------- 4) grouped bf16 WMMA GEMM --------------------------------
// 128x128 block tile, 8 waves, wave tile 32x64 -> 2x4 v_wmma_f32_16x16x32_bf16
// LDS double-buffered, one barrier per k-step.
// BPRE 1: B pre-transposed bf16 [NDIM][KDIM]; tiles staged with
//         global_load_async_to_lds_b128 (ASYNCcnt), no VGPR round-trip.
// BPRE 0: B fp32 [KDIM][NDIM], register-staged + converted (fallback).
// MODE 0: out = relu(A@W)^2 -> bf16 OutBf (hidden)
// MODE 1: out = A@W -> fp32 OutF (shared rows) / bf16 OutBf (expert slot rows)
template <int KDIM, int NDIM, int MODE, bool BPRE>
__global__ __launch_bounds__(256) void gemm_kernel(
    const __bf16* __restrict__ A,
    const void* __restrict__ Bsh,
    const void* __restrict__ Bex,
    __bf16* __restrict__ OutBf,
    float*  __restrict__ OutF)
{
  __shared__ alignas(16) __bf16 sA[2][128][40];   // [buf][m][k], padded
  __shared__ alignas(16) __bf16 sB[2][128][40];   // [buf][n][k], padded
  constexpr unsigned LBUF = 128 * 40 * 2;         // bytes per buffer layer

  const int row0 = blockIdx.y * 128;
  const int n0   = blockIdx.x * 128;
  const size_t goff = (row0 < N_) ? 0
      : (size_t)((row0 - N_) / CAP_) * KDIM * NDIM;
  const __bf16* Wb = nullptr;
  const float*  Wf = nullptr;
  if constexpr (BPRE) Wb = (row0 < N_) ? (const __bf16*)Bsh : ((const __bf16*)Bex + goff);
  else                Wf = (row0 < N_) ? (const float*)Bsh  : ((const float*)Bex + goff);

  const int tid  = threadIdx.x;
  const int wave = tid >> 5;
  const int lane = tid & 31;
  const int wm = wave >> 1, wn = wave & 1;      // wave tile: rows 32*wm, cols 64*wn
  const int lr = lane & 15, lh = lane >> 4;

  const int lar = tid >> 1, lac = (tid & 1) * 16;   // A/B(pre) loader: 128 x 32
  const int fkr = tid >> 3, fcc = (tid & 7) * 16;   // B(f32) loader: 32 x 128

  v8f acc[2][4] = {};
  const int KSTEPS = KDIM / 32;

  // ---- fast path: async global->LDS staging -------------------------------
  unsigned laA = 0, laB = 0;
  const __bf16 *gA = nullptr, *gB = nullptr;
  // ---- fallback: register staging ----------------------------------------
  bf8 r0, r1, r2, r3;
  float4 bF[4];

  auto fetch = [&](int k0) {        // fallback only
    const __bf16* pa = A + (size_t)(row0 + lar) * KDIM + k0 + lac;
    r0 = *(const bf8*)pa;
    r1 = *(const bf8*)(pa + 8);
    const float* pb = Wf + (size_t)(k0 + fkr) * NDIM + n0 + fcc;
    #pragma unroll
    for (int q = 0; q < 4; ++q) bF[q] = *(const float4*)(pb + q * 4);
  };
  auto commit = [&](int buf) {      // fallback only
    *(bf8*)(&sA[buf][lar][lac])     = r0;
    *(bf8*)(&sA[buf][lar][lac + 8]) = r1;
    #pragma unroll
    for (int q = 0; q < 4; ++q) {
      sB[buf][fcc + q * 4 + 0][fkr] = f2bf(bF[q].x);
      sB[buf][fcc + q * 4 + 1][fkr] = f2bf(bF[q].y);
      sB[buf][fcc + q * 4 + 2][fkr] = f2bf(bF[q].z);
      sB[buf][fcc + q * 4 + 3][fkr] = f2bf(bF[q].w);
    }
  };
  auto issue_async = [&](int k0, int buf) {   // fast path only
    const unsigned da = laA + (unsigned)buf * LBUF;
    const unsigned db = laB + (unsigned)buf * LBUF;
    async_copy_b128    (da, gA + k0);
    async_copy_b128_o16(da, gA + k0);
    async_copy_b128    (db, gB + k0);
    async_copy_b128_o16(db, gB + k0);
  };

  if constexpr (BPRE) {
    laA = (unsigned)(size_t)&sA[0][lar][lac];   // LDS flat addr low 32 = offset
    laB = (unsigned)(size_t)&sB[0][lar][lac];
    gA = A  + (size_t)(row0 + lar) * KDIM + lac;
    gB = Wb + (size_t)(n0  + lar) * KDIM + lac;
    issue_async(0, 0);
  } else {
    fetch(0);
    commit(0);
  }

  for (int ks = 0; ks < KSTEPS; ++ks) {
    if constexpr (BPRE) wait_async0();   // this wave's LDS writes retired
    __syncthreads();
    const int cur = ks & 1;
    if (ks + 1 < KSTEPS) {
      if constexpr (BPRE) issue_async((ks + 1) * 32, 1 - cur);
      else fetch((ks + 1) * 32);
    }
    if constexpr (!BPRE) {
      if (ks + 2 < KSTEPS) {
        const int kp = (ks + 2) * 32;
        __builtin_prefetch(A + (size_t)(row0 + lar) * KDIM + kp + lac, 0, 1);
        __builtin_prefetch(Wf + (size_t)(kp + fkr) * NDIM + n0 + fcc, 0, 1);
      }
    }

    // A fragments (ISA 16-bit layout: lane(lr,lh) -> K = lh*8+{0..7}, +16)
    v16bf af[2];
    #pragma unroll
    for (int mt = 0; mt < 2; ++mt) {
      const int r = 32 * wm + 16 * mt + lr;
      const bf8 lo = *(const bf8*)(&sA[cur][r][lh * 8]);
      const bf8 hi = *(const bf8*)(&sA[cur][r][lh * 8 + 16]);
      #pragma unroll
      for (int i = 0; i < 8; ++i) { af[mt][i] = lo[i]; af[mt][8 + i] = hi[i]; }
    }
    #pragma unroll
    for (int nt = 0; nt < 4; ++nt) {
      const int c = 64 * wn + 16 * nt + lr;
      const bf8 lo = *(const bf8*)(&sB[cur][c][lh * 8]);
      const bf8 hi = *(const bf8*)(&sB[cur][c][lh * 8 + 16]);
      v16bf bfr;
      #pragma unroll
      for (int i = 0; i < 8; ++i) { bfr[i] = lo[i]; bfr[8 + i] = hi[i]; }
      #pragma unroll
      for (int mt = 0; mt < 2; ++mt)
        acc[mt][nt] = __builtin_amdgcn_wmma_f32_16x16x32_bf16(
            false, af[mt], false, bfr, (short)0, acc[mt][nt], false, false);
    }

    if constexpr (!BPRE) {
      if (ks + 1 < KSTEPS) commit(1 - cur);
    }
  }

  // epilogue: C/D layout — VGPR r, lane(lr,lh) -> (M = 8*lh + r, N = lr)
  #pragma unroll
  for (int mt = 0; mt < 2; ++mt)
    #pragma unroll
    for (int nt = 0; nt < 4; ++nt)
      #pragma unroll
      for (int r = 0; r < 8; ++r) {
        const int row = row0 + 32 * wm + 16 * mt + 8 * lh + r;
        const int col = n0 + 64 * wn + 16 * nt + lr;
        float v = acc[mt][nt][r];
        if (MODE == 0) {
          v = fmaxf(v, 0.f); v = v * v;                    // relu^2
          OutBf[(size_t)row * NDIM + col] = f2bf(v);
        } else {
          if (row0 < N_) OutF[(size_t)row * NDIM + col] = v;        // shared -> y
          else OutBf[(size_t)(row - N_) * NDIM + col] = f2bf(v);    // expert slots
        }
      }
}

// ---------------- 5) combine: y += sum_k gate_k * expert_out[slot_k] -------
__global__ __launch_bounds__(256) void combine_kernel(
    float* __restrict__ y, const __bf16* __restrict__ Omoe,
    const int* __restrict__ slot_of, const float* __restrict__ gate)
{
  const int n = blockIdx.x;
  const int c = threadIdx.x * 4;
  float4 acc = *(const float4*)(y + (size_t)n * D_ + c);
  #pragma unroll
  for (int k = 0; k < KTOP; ++k) {
    const int s = slot_of[n * 2 + k];
    if (s >= 0) {
      const float g = gate[n * 2 + k];
      const bf4 v = *(const bf4*)(Omoe + (size_t)s * D_ + c);
      acc.x += g * bf2f(v[0]); acc.y += g * bf2f(v[1]);
      acc.z += g * bf2f(v[2]); acc.w += g * bf2f(v[3]);
    }
  }
  *(float4*)(y + (size_t)n * D_ + c) = acc;
}

// ---------------- 6) aux loss: 0.01 * E * sum_e f_e * p_e ------------------
__global__ __launch_bounds__(256) void aux_kernel(
    const float* __restrict__ probs, const int* __restrict__ top1,
    float* __restrict__ out_aux)
{
  __shared__ float sp[256];
  __shared__ float sf[256];
  const int t = threadIdx.x;
  const int e = t & 15, j = t >> 4;
  float ps = 0.f, fs = 0.f;
  for (int n = j; n < N_; n += 16) {
    ps += probs[(size_t)n * E_ + e];
    fs += (top1[n] == e) ? 1.f : 0.f;
  }
  sp[t] = ps; sf[t] = fs;
  __syncthreads();
  if (t < 16) {
    float P = 0.f, F = 0.f;
    for (int q = 0; q < 16; ++q) { P += sp[q * 16 + t]; F += sf[q * 16 + t]; }
    sp[t] = (F / (float)N_) * (P / (float)N_);
  }
  __syncthreads();
  if (t == 0) {
    float a = 0.f;
    for (int q = 0; q < 16; ++q) a += sp[q];
    *out_aux = 0.01f * (float)E_ * a;
  }
}

// ---------------- launch ---------------------------------------------------
extern "C" void kernel_launch(void* const* d_in, const int* in_sizes, int n_in,
                              void* d_out, int out_size, void* d_ws, size_t ws_size,
                              hipStream_t stream)
{
  (void)in_sizes; (void)n_in; (void)out_size;
  const float* x    = (const float*)d_in[0];
  const float* rw   = (const float*)d_in[1];
  const float* wfc  = (const float*)d_in[2];
  const float* wpj  = (const float*)d_in[3];
  const float* wsfc = (const float*)d_in[4];
  const float* wspj = (const float*)d_in[5];
  float* y = (float*)d_out;

  char* ws = (char*)d_ws;
  size_t off = 0;
  auto alloc = [&](size_t bytes) -> void* {
    void* p = ws + off;
    off += (bytes + 255) & ~(size_t)255;
    return p;
  };
  float*  probs      = (float*) alloc((size_t)N_ * E_ * 4);
  int*    top1       = (int*)   alloc((size_t)N_ * 4);
  int*    eidx       = (int*)   alloc((size_t)N_ * KTOP * 4);
  float*  gate       = (float*) alloc((size_t)N_ * KTOP * 4);
  int*    slot_of    = (int*)   alloc((size_t)N_ * KTOP * 4);
  int*    slot_token = (int*)   alloc((size_t)E_ * CAP_ * 4);
  __bf16* Abuf       = (__bf16*)alloc((size_t)MROWS_ * D_ * 2);
  __bf16* Hbuf       = (__bf16*)alloc((size_t)MROWS_ * H_ * 2);
  __bf16* Omoe       = (__bf16*)alloc((size_t)E_ * CAP_ * D_ * 2);
  // pre-transposed bf16 weights (fast path), allocated last for ws_size gate
  __bf16* Wtfc  = (__bf16*)alloc((size_t)E_ * D_ * H_ * 2);   // [E][H][D]
  __bf16* Wtpj  = (__bf16*)alloc((size_t)E_ * H_ * D_ * 2);   // [E][D][H]
  __bf16* Wtsfc = (__bf16*)alloc((size_t)D_ * H_ * 2);        // [H][D]
  __bf16* Wtspj = (__bf16*)alloc((size_t)H_ * D_ * 2);        // [D][H]
  const bool pre = (off <= ws_size);   // deterministic path selection

  router_kernel<<<N_ / 8, 256, 0, stream>>>(x, rw, probs, top1, eidx, gate);
  dispatch_kernel<<<1, 256, 0, stream>>>(eidx, slot_of, slot_token);
  gather_kernel<<<MROWS_, 256, 0, stream>>>(x, slot_token, Abuf);

  const dim3 g1(H_ / 128, MROWS_ / 128);
  const dim3 g2(D_ / 128, MROWS_ / 128);
  if (pre) {
    wtrans_kernel<<<dim3(D_ / 32, H_ / 32, E_), 256, 0, stream>>>(wfc,  Wtfc,  D_, H_);
    wtrans_kernel<<<dim3(H_ / 32, D_ / 32, E_), 256, 0, stream>>>(wpj,  Wtpj,  H_, D_);
    wtrans_kernel<<<dim3(D_ / 32, H_ / 32, 1),  256, 0, stream>>>(wsfc, Wtsfc, D_, H_);
    wtrans_kernel<<<dim3(H_ / 32, D_ / 32, 1),  256, 0, stream>>>(wspj, Wtspj, H_, D_);
    gemm_kernel<D_, H_, 0, true><<<g1, 256, 0, stream>>>(Abuf, Wtsfc, Wtfc, Hbuf, nullptr);
    gemm_kernel<H_, D_, 1, true><<<g2, 256, 0, stream>>>(Hbuf, Wtspj, Wtpj, Omoe, y);
  } else {
    gemm_kernel<D_, H_, 0, false><<<g1, 256, 0, stream>>>(Abuf, wsfc, wfc, Hbuf, nullptr);
    gemm_kernel<H_, D_, 1, false><<<g2, 256, 0, stream>>>(Hbuf, wspj, wpj, Omoe, y);
  }

  combine_kernel<<<N_, 256, 0, stream>>>(y, Omoe, slot_of, gate);
  aux_kernel<<<1, 256, 0, stream>>>(probs, top1, y + (size_t)N_ * D_);
}
